// SwinTransformerLayer_27470610826082
// MI455X (gfx1250) — compile-verified, matrix operand order checked by reading, not run on hardware
//
#include <hip/hip_runtime.h>
#include <cstddef>

typedef __attribute__((ext_vector_type(16))) __bf16 v16bf;
typedef __attribute__((ext_vector_type(8)))  __bf16 v8bf;
typedef __attribute__((ext_vector_type(8)))  float  v8f;

#define LNS 136   // 64x128 bf16 tile row stride (pad 8 -> conflict-free frag loads)
#define VTS 72    // vT (128 x 64) row stride
#define PSS 72    // P (64 x 64) row stride
#define HSS 520   // MLP hidden (64 x 512) row stride

__device__ __forceinline__ v8f wmma_bf16(v16bf a, v16bf b, v8f c) {
  return __builtin_amdgcn_wmma_f32_16x16x32_bf16(false, a, false, b, (short)0, c, false, false);
}

// Load a 16x32 A-style fragment (also used for B given an N-major matrix).
// Lane L: row = rbase + (L&15); K group g = L>>4 -> K = kbase + g*8 + {0..7} and kbase + 16 + g*8 + {0..7}.
__device__ __forceinline__ v16bf ldfrag(const __bf16* mat, int stride, int rbase, int kbase) {
  int lane = threadIdx.x & 31;
  const __bf16* p = mat + (rbase + (lane & 15)) * stride + kbase + ((lane >> 4) << 3);
  v8bf lo = *(const v8bf*)p;
  v8bf hi = *(const v8bf*)(p + 16);
  return __builtin_shufflevector(lo, hi, 0,1,2,3,4,5,6,7,8,9,10,11,12,13,14,15);
}

__device__ __forceinline__ float half_red_add(float v) {
  v += __shfl_xor(v, 1, 32); v += __shfl_xor(v, 2, 32);
  v += __shfl_xor(v, 4, 32); v += __shfl_xor(v, 8, 32);
  return v;
}
__device__ __forceinline__ float half_red_max(float v) {
  v = fmaxf(v, __shfl_xor(v, 1, 32)); v = fmaxf(v, __shfl_xor(v, 2, 32));
  v = fmaxf(v, __shfl_xor(v, 4, 32)); v = fmaxf(v, __shfl_xor(v, 8, 32));
  return v;
}
__device__ __forceinline__ float wave_red_add(float v) {
  v += __shfl_xor(v, 16, 32);
  return half_red_add(v);
}

// ---------------- prep: bf16 N-major weights + relative position bias (4,64,64) ---------
__global__ __launch_bounds__(256) void prep_kernel(
    const float* __restrict__ qkv_w, const float* __restrict__ proj_w,
    const float* __restrict__ w1, const float* __restrict__ w2,
    const float* __restrict__ rpb,
    __bf16* __restrict__ qkvT, __bf16* __restrict__ projT,
    __bf16* __restrict__ w1T, __bf16* __restrict__ w2T,
    float* __restrict__ bias4) {
  int i = blockIdx.x * 256 + threadIdx.x;  // 65536 threads
  if (i < 384 * 128) { int n = i >> 7, k = i & 127; qkvT[i] = (__bf16)qkv_w[k * 384 + n]; }
  if (i < 128 * 128) { int n = i >> 7, k = i & 127; projT[i] = (__bf16)proj_w[k * 128 + n]; }
  { int n = i >> 7, k = i & 127; w1T[i] = (__bf16)w1[k * 512 + n]; }   // 512x128
  { int n = i >> 9, k = i & 511; w2T[i] = (__bf16)w2[k * 128 + n]; }   // 128x512
  if (i < 4 * 64 * 64) {
    int h = i >> 12, r = i & 4095, ii = r >> 6, jj = r & 63;
    int dy = (ii >> 3) - (jj >> 3) + 7, dx = (ii & 7) - (jj & 7) + 7;
    bias4[i] = rpb[(dy * 15 + dx) * 4 + h];
  }
}

// ---------------- fused LN1 + shift + QKV + attention + proj + residual (one block/window) -----
__global__ __launch_bounds__(256) void attn_kernel(
    const float* __restrict__ x, const float* __restrict__ ln1_g, const float* __restrict__ ln1_b,
    const float* __restrict__ qkv_b, const float* __restrict__ proj_b,
    const __bf16* __restrict__ qkvT, const __bf16* __restrict__ projT,
    const float* __restrict__ bias4, float* __restrict__ x1) {
  extern __shared__ __align__(16) char smem[];
  __bf16* lnX = (__bf16*)smem;          // 64 x LNS (LN'd input; reused as attention output)
  __bf16* qS  = lnX + 64 * LNS;         // 64 x LNS (q, pre-scaled)
  __bf16* kS  = qS + 64 * LNS;          // 64 x LNS
  __bf16* vTS = kS + 64 * LNS;          // 128 x VTS (v transposed: [head*32+hd][token])
  __bf16* PS  = vTS + 128 * VTS;        // 4 * 64 x PSS (softmax probs per head)

  const int tid = threadIdx.x, wv = tid >> 5, lane = tid & 31;
  const int cg = lane & 15, g = lane >> 4;
  const int ww = blockIdx.x, b = ww >> 10, w = ww & 1023, wy = w >> 5, wx = w & 31;
  const int gy0 = wy * 8, gx0 = wx * 8;

  // ---- LN1 with cyclic shift gather; one wave handles 8 tokens, 4 ch/lane ----
  {
    const float4 g4 = ((const float4*)ln1_g)[lane];
    const float4 b4 = ((const float4*)ln1_b)[lane];
    for (int r = 0; r < 8; r++) {
      int t = wv * 8 + r;
      int gy = (gy0 + (t >> 3) + 4) & 255, gx = (gx0 + (t & 7) + 4) & 255;
      const float* px = x + ((((size_t)b * 256 + gy) * 256 + gx) << 7);
      float4 v = ((const float4*)px)[lane];
      float s  = v.x + v.y + v.z + v.w;
      float s2 = v.x * v.x + v.y * v.y + v.z * v.z + v.w * v.w;
      s = wave_red_add(s); s2 = wave_red_add(s2);
      float mean = s * (1.f / 128.f);
      float var  = s2 * (1.f / 128.f) - mean * mean;
      float rstd = rsqrtf(var + 1e-3f);
      __bf16* o = lnX + t * LNS + lane * 4;
      o[0] = (__bf16)((v.x - mean) * rstd * g4.x + b4.x);
      o[1] = (__bf16)((v.y - mean) * rstd * g4.y + b4.y);
      o[2] = (__bf16)((v.z - mean) * rstd * g4.z + b4.z);
      o[3] = (__bf16)((v.w - mean) * rstd * g4.w + b4.w);
    }
  }
  __syncthreads();

  // ---- QKV GEMM: M=64, N=384, K=128; wave owns 3 N-tiles ----
  const float qscale = 0.08838834764831845f;  // 128^-0.5
  for (int nn = 0; nn < 3; nn++) {
    int nt = wv * 3 + nn;
    v8f acc[4] = {};
    for (int kf = 0; kf < 4; kf++) {
      v16bf bf = ldfrag(qkvT, 128, nt * 16, kf * 32);
#pragma unroll
      for (int mt = 0; mt < 4; mt++)
        acc[mt] = wmma_bf16(ldfrag(lnX, LNS, mt * 16, kf * 32), bf, acc[mt]);
    }
    int n = nt * 16 + cg;
    float bb = qkv_b[n];
    int part = nt >> 3;     // 0:q 1:k 2:v (tile never crosses a 128 boundary)
    int f = n & 127;
#pragma unroll
    for (int mt = 0; mt < 4; mt++)
#pragma unroll
      for (int j = 0; j < 8; j++) {
        int t = mt * 16 + j + 8 * g;
        float val = acc[mt][j] + bb;
        if (part == 0)      qS[t * LNS + f]  = (__bf16)(val * qscale);
        else if (part == 1) kS[t * LNS + f]  = (__bf16)val;
        else                vTS[f * VTS + t] = (__bf16)val;
      }
  }
  __syncthreads();

  // ---- attention: wave = (head wv>>1, row-half (wv&1)*32) ----
  {
    const int h = wv >> 1, rbase = (wv & 1) * 32;
    auto region = [&](int t) -> int {
      int yy = gy0 + (t >> 3), xx = gx0 + (t & 7);
      int ry = yy < 248 ? 0 : (yy < 252 ? 1 : 2);
      int rx = xx < 248 ? 0 : (xx < 252 ? 1 : 2);
      return ry * 3 + rx;
    };
    for (int s = 0; s < 2; s++) {
      int mrow = rbase + s * 16;
      v16bf a = ldfrag(qS, LNS, mrow, h * 32);
      v8f sc[4];
#pragma unroll
      for (int nt = 0; nt < 4; nt++) {
        v8f z = {};
        sc[nt] = wmma_bf16(a, ldfrag(kS, LNS, nt * 16, h * 32), z);
      }
#pragma unroll
      for (int j = 0; j < 8; j++) {
        int i = mrow + j + 8 * g;
        int regi = region(i);
        float mx = -1e30f;
#pragma unroll
        for (int nt = 0; nt < 4; nt++) {
          int c = nt * 16 + cg;
          float bm = bias4[(h * 64 + i) * 64 + c];
          if (region(c) != regi) bm -= 100.f;
          float v = sc[nt][j] + bm;
          sc[nt][j] = v;
          mx = fmaxf(mx, v);
        }
        mx = half_red_max(mx);
        float sum = 0.f;
#pragma unroll
        for (int nt = 0; nt < 4; nt++) {
          float e = __expf(sc[nt][j] - mx);
          sc[nt][j] = e; sum += e;
        }
        sum = half_red_add(sum);
        float rs = 1.f / sum;
#pragma unroll
        for (int nt = 0; nt < 4; nt++)
          PS[(h * 64 + i) * PSS + nt * 16 + cg] = (__bf16)(sc[nt][j] * rs);
      }
      // P @ v : M=16 strip, N=32 (head dim), K=64
      v16bf p0 = ldfrag(PS + h * 64 * PSS, PSS, mrow, 0);
      v16bf p1 = ldfrag(PS + h * 64 * PSS, PSS, mrow, 32);
#pragma unroll
      for (int nt2 = 0; nt2 < 2; nt2++) {
        v8f o = {};
        o = wmma_bf16(p0, ldfrag(vTS, VTS, h * 32 + nt2 * 16, 0), o);
        o = wmma_bf16(p1, ldfrag(vTS, VTS, h * 32 + nt2 * 16, 32), o);
#pragma unroll
        for (int j = 0; j < 8; j++)
          lnX[(mrow + j + 8 * g) * LNS + h * 32 + nt2 * 16 + cg] = (__bf16)o[j];
      }
    }
  }
  __syncthreads();

  // ---- proj GEMM + shift-back residual: M=64, N=128, K=128; wave owns 1 N-tile ----
  {
    v8f acc[4] = {};
    for (int kf = 0; kf < 4; kf++) {
      v16bf bf = ldfrag(projT, 128, wv * 16, kf * 32);
#pragma unroll
      for (int mt = 0; mt < 4; mt++)
        acc[mt] = wmma_bf16(ldfrag(lnX, LNS, mt * 16, kf * 32), bf, acc[mt]);
    }
    int ch = wv * 16 + cg;
    float pb = proj_b[ch];
#pragma unroll
    for (int mt = 0; mt < 4; mt++)
#pragma unroll
      for (int j = 0; j < 8; j++) {
        int t = mt * 16 + j + 8 * g;
        int gy = (gy0 + (t >> 3) + 4) & 255, gx = (gx0 + (t & 7) + 4) & 255;
        size_t idx = ((((size_t)b * 256 + gy) * 256 + gx) << 7) + ch;
        x1[idx] = x[idx] + acc[mt][j] + pb;
      }
  }
}

// ---------------- fused LN2 + MLP + residual (one block / 64 tokens) ----------------
__global__ __launch_bounds__(256) void mlp_kernel(
    const float* __restrict__ x1, const float* __restrict__ ln2_g, const float* __restrict__ ln2_b,
    const float* __restrict__ mlp_b1, const float* __restrict__ mlp_b2,
    const __bf16* __restrict__ w1T, const __bf16* __restrict__ w2T,
    float* __restrict__ out) {
  extern __shared__ __align__(16) char smem[];
  __bf16* lnX = (__bf16*)smem;     // 64 x LNS
  __bf16* HS  = lnX + 64 * LNS;    // 64 x HSS (gelu hidden)

  const int tid = threadIdx.x, wv = tid >> 5, lane = tid & 31;
  const int cg = lane & 15, g = lane >> 4;
  const size_t tbase = (size_t)blockIdx.x * 64;

  {
    const float4 g4 = ((const float4*)ln2_g)[lane];
    const float4 b4 = ((const float4*)ln2_b)[lane];
    for (int r = 0; r < 8; r++) {
      int t = wv * 8 + r;
      const float* px = x1 + ((tbase + t) << 7);
      float4 v = ((const float4*)px)[lane];
      float s  = v.x + v.y + v.z + v.w;
      float s2 = v.x * v.x + v.y * v.y + v.z * v.z + v.w * v.w;
      s = wave_red_add(s); s2 = wave_red_add(s2);
      float mean = s * (1.f / 128.f);
      float var  = s2 * (1.f / 128.f) - mean * mean;
      float rstd = rsqrtf(var + 1e-3f);
      __bf16* o = lnX + t * LNS + lane * 4;
      o[0] = (__bf16)((v.x - mean) * rstd * g4.x + b4.x);
      o[1] = (__bf16)((v.y - mean) * rstd * g4.y + b4.y);
      o[2] = (__bf16)((v.z - mean) * rstd * g4.z + b4.z);
      o[3] = (__bf16)((v.w - mean) * rstd * g4.w + b4.w);
    }
  }
  __syncthreads();

  // GEMM1 + exact GELU: M=64, N=512, K=128; wave owns 4 N-tiles in 2 passes
  for (int p = 0; p < 2; p++) {
    v8f acc[4][2] = {};
    int n0 = wv * 4 + p * 2;
    for (int kf = 0; kf < 4; kf++) {
      v16bf bf0 = ldfrag(w1T, 128, n0 * 16, kf * 32);
      v16bf bf1 = ldfrag(w1T, 128, (n0 + 1) * 16, kf * 32);
#pragma unroll
      for (int mt = 0; mt < 4; mt++) {
        v16bf a = ldfrag(lnX, LNS, mt * 16, kf * 32);
        acc[mt][0] = wmma_bf16(a, bf0, acc[mt][0]);
        acc[mt][1] = wmma_bf16(a, bf1, acc[mt][1]);
      }
    }
#pragma unroll
    for (int q = 0; q < 2; q++) {
      int n = (n0 + q) * 16 + cg;
      float bb = mlp_b1[n];
#pragma unroll
      for (int mt = 0; mt < 4; mt++)
#pragma unroll
        for (int j = 0; j < 8; j++) {
          float v = acc[mt][q][j] + bb;
          v = 0.5f * v * (1.f + erff(v * 0.70710678118654752f));
          HS[(mt * 16 + j + 8 * g) * HSS + n] = (__bf16)v;
        }
    }
  }
  __syncthreads();

  // GEMM2 + residual: M=64, N=128, K=512; wave owns 1 N-tile
  {
    v8f acc[4] = {};
    for (int kf = 0; kf < 16; kf++) {
      v16bf bf = ldfrag(w2T, 512, wv * 16, kf * 32);
#pragma unroll
      for (int mt = 0; mt < 4; mt++)
        acc[mt] = wmma_bf16(ldfrag(HS, HSS, mt * 16, kf * 32), bf, acc[mt]);
    }
    int ch = wv * 16 + cg;
    float bb = mlp_b2[ch];
#pragma unroll
    for (int mt = 0; mt < 4; mt++)
#pragma unroll
      for (int j = 0; j < 8; j++) {
        size_t idx = ((tbase + mt * 16 + j + 8 * g) << 7) + ch;
        out[idx] = x1[idx] + acc[mt][j] + bb;
      }
  }
}

extern "C" void kernel_launch(void* const* d_in, const int* in_sizes, int n_in,
                              void* d_out, int out_size, void* d_ws, size_t ws_size,
                              hipStream_t stream) {
  const float* x      = (const float*)d_in[0];
  const float* ln1_g  = (const float*)d_in[1];
  const float* ln1_b  = (const float*)d_in[2];
  const float* qkv_w  = (const float*)d_in[3];
  const float* qkv_b  = (const float*)d_in[4];
  const float* rpb    = (const float*)d_in[5];
  const float* proj_w = (const float*)d_in[6];
  const float* proj_b = (const float*)d_in[7];
  const float* ln2_g  = (const float*)d_in[8];
  const float* ln2_b  = (const float*)d_in[9];
  const float* w1     = (const float*)d_in[10];
  const float* b1     = (const float*)d_in[11];
  const float* w2     = (const float*)d_in[12];
  const float* b2     = (const float*)d_in[13];
  float* out = (float*)d_out;

  char* p = (char*)d_ws;
  __bf16* qkvT  = (__bf16*)p; p += 384 * 128 * 2;
  __bf16* projT = (__bf16*)p; p += 128 * 128 * 2;
  __bf16* w1T   = (__bf16*)p; p += 512 * 128 * 2;
  __bf16* w2T   = (__bf16*)p; p += 128 * 512 * 2;
  float*  bias4 = (float*)p;  p += 4 * 64 * 64 * 4;
  float*  x1    = (float*)p;                          // 4*256*256*128 f32 = 134 MB

  prep_kernel<<<256, 256, 0, stream>>>(qkv_w, proj_w, w1, w2, rpb, qkvT, projT, w1T, w2T, bias4);

  size_t sm2 = (size_t)(64 * LNS * 3 + 128 * VTS + 4 * 64 * PSS) * sizeof(__bf16); // 107520 B
  attn_kernel<<<4096, 256, sm2, stream>>>(x, ln1_g, ln1_b, qkv_b, proj_b, qkvT, projT, bias4, x1);

  size_t sm3 = (size_t)(64 * LNS + 64 * HSS) * sizeof(__bf16);                     // 83968 B
  mlp_kernel<<<4096, 256, sm3, stream>>>(x1, ln2_g, ln2_b, b1, b2, w1T, w2T, out);
}
